// GCN_GRU_QoR_37795712205132
// MI455X (gfx1250) — matrix operand order repaired and checked by reading, not compiled, not run on hardware
//
#include <hip/hip_runtime.h>
#include <hip/hip_bf16.h>

typedef __attribute__((ext_vector_type(16))) _Float16 v16h;
typedef __attribute__((ext_vector_type(8)))  float    v8f;

#define GCNH 64
#define GRUH 128
#define TSTEPS 8
#define SCAN_B 256

// ---------------------------------------------------------------------------
// CSR build step 1: in-degree counts (int atomics, once per call)
// ---------------------------------------------------------------------------
__global__ void count_kernel(const long long* __restrict__ dst, int* __restrict__ counts, int E) {
    int e = blockIdx.x * blockDim.x + threadIdx.x;
    if (e < E) atomicAdd(&counts[dst[e]], 1);
}

// dinv[n] = rsqrt(in_degree + 1)  (self-loop)
__global__ void dinv_kernel(const int* __restrict__ counts, float* __restrict__ dinv, int n) {
    int i = blockIdx.x * blockDim.x + threadIdx.x;
    if (i < n) dinv[i] = rsqrtf((float)counts[i] + 1.0f);
}

// ---------------------------------------------------------------------------
// 3-kernel exclusive scan of counts -> offsets (N <= SCAN_B*SCAN_B)
// ---------------------------------------------------------------------------
__global__ void scan1_kernel(const int* __restrict__ counts, int* __restrict__ scanned,
                             int* __restrict__ blockSums, int n) {
    __shared__ int tmp[SCAN_B];
    int i = blockIdx.x * SCAN_B + threadIdx.x;
    int v = (i < n) ? counts[i] : 0;
    tmp[threadIdx.x] = v;
    __syncthreads();
    for (int off = 1; off < SCAN_B; off <<= 1) {
        int t = (threadIdx.x >= off) ? tmp[threadIdx.x - off] : 0;
        __syncthreads();
        tmp[threadIdx.x] += t;
        __syncthreads();
    }
    if (i < n) scanned[i] = tmp[threadIdx.x] - v;                 // exclusive
    if (threadIdx.x == SCAN_B - 1) blockSums[blockIdx.x] = tmp[threadIdx.x];
}

__global__ void scan2_kernel(int* __restrict__ bs, int nb) {      // single block
    __shared__ int tmp[SCAN_B];
    int tid = threadIdx.x;
    int v = (tid < nb) ? bs[tid] : 0;
    tmp[tid] = v;
    __syncthreads();
    for (int off = 1; off < SCAN_B; off <<= 1) {
        int t = (tid >= off) ? tmp[tid - off] : 0;
        __syncthreads();
        tmp[tid] += t;
        __syncthreads();
    }
    if (tid < nb) bs[tid] = tmp[tid] - v;                         // exclusive
}

__global__ void scan3_kernel(int* __restrict__ scanned, const int* __restrict__ bs, int n) {
    int i = blockIdx.x * SCAN_B + threadIdx.x;
    if (i < n) scanned[i] += bs[blockIdx.x];
}

__global__ void copy_kernel(const int* __restrict__ a, int* __restrict__ b, int n) {
    int i = blockIdx.x * blockDim.x + threadIdx.x;
    if (i < n) b[i] = a[i];
}

// ---------------------------------------------------------------------------
// CSR build step 2: fill {src, w} records. w = dinv[src]*dinv[dst].
// ---------------------------------------------------------------------------
__global__ void fill_kernel(const long long* __restrict__ eidx, const float* __restrict__ dinv,
                            int* __restrict__ cursor, int2* __restrict__ csr, int E) {
    int e = blockIdx.x * blockDim.x + threadIdx.x;
    if (e >= E) return;
    int s = (int)eidx[e];
    int d = (int)eidx[(long long)E + e];
    float w = dinv[s] * dinv[d];
    int pos = atomicAdd(&cursor[d], 1);
    csr[pos] = make_int2(s, __float_as_int(w));
}

// ---------------------------------------------------------------------------
// Layer-1 aggregation on RAW 3-dim features (exact rank-3 trick):
// aggx[n] = dinv^2 * x[n] + sum_{(s,w) in CSR[n]} w * x[s]
// One thread per node, zero atomics.
// ---------------------------------------------------------------------------
__global__ void gather3_kernel(const int* __restrict__ offsets, const int* __restrict__ counts,
                               const int2* __restrict__ csr, const float* __restrict__ x,
                               const float* __restrict__ dinv, float* __restrict__ aggx, int n) {
    int node = blockIdx.x * blockDim.x + threadIdx.x;
    if (node >= n) return;
    float di = dinv[node];
    float sl = di * di;
    const float* xs = x + (long long)node * 3;
    float a0 = sl * xs[0], a1 = sl * xs[1], a2 = sl * xs[2];
    int start = offsets[node];
    int cnt   = counts[node];
    for (int k = 0; k < cnt; ++k) {
        int2 ew = csr[start + k];
        const float* xv = x + (long long)ew.x * 3;
        float w = __int_as_float(ew.y);
        a0 += w * xv[0];
        a1 += w * xv[1];
        a2 += w * xv[2];
    }
    aggx[(long long)node * 3 + 0] = a0;
    aggx[(long long)node * 3 + 1] = a1;
    aggx[(long long)node * 3 + 2] = a2;
}

// ---------------------------------------------------------------------------
// Layer-1 epilogue: h1 = relu( aggx @ W1 + b1 )
// ---------------------------------------------------------------------------
__global__ void epi1_kernel(const float* __restrict__ aggx, const float* __restrict__ W1,
                            const float* __restrict__ b1, float* __restrict__ h, int n) {
    int i = blockIdx.x * blockDim.x + threadIdx.x;   // over N*64
    if (i >= n * GCNH) return;
    int node = i >> 6;
    int f    = i & 63;
    const float* ax = aggx + (long long)node * 3;
    float v = ax[0] * W1[f] + ax[1] * W1[GCNH + f] + ax[2] * W1[2 * GCNH + f] + b1[f];
    h[i] = fmaxf(v, 0.0f);
}

// ---------------------------------------------------------------------------
// WMMA GEMM: C[N,64] = A[N,64] @ W[64,64], one wave per 16x16 tile,
// v_wmma_f32_16x16x32_f16, f32 accumulate. N must be a multiple of 16.
// ---------------------------------------------------------------------------
__global__ void gemm64_wmma_kernel(const float* __restrict__ A,
                                   const float* __restrict__ W,
                                   float* __restrict__ C, int nrows) {
    const int waves_per_blk = blockDim.x >> 5;
    int wave = blockIdx.x * waves_per_blk + (threadIdx.x >> 5);
    int lane = threadIdx.x & 31;
    const int tiles_n = GCNH / 16;                 // 4
    int total_tiles = (nrows / 16) * tiles_n;
    if (wave >= total_tiles) return;               // uniform per wave: EXEC stays all-1s
    int tile_m = wave / tiles_n;
    int tile_n = wave - tile_m * tiles_n;
    int m  = lane & 15;
    int hi = lane >> 4;
    int row  = tile_m * 16 + m;
    int col0 = tile_n * 16;
    int n    = lane & 15;

    v8f c = {};
    const float* arow = A + (long long)row * GCNH;
    #pragma unroll
    for (int kk = 0; kk < 2; ++kk) {               // K blocks of 32
        int kbase = kk * 32;
        v16h a, b;
        #pragma unroll
        for (int e = 0; e < 16; ++e) {
            int ka = hi * 16 + e;                      // 16-bit A 16x32 layout (contiguous k, matches B)
            a[e] = (_Float16)arow[kbase + ka];
        }
        #pragma unroll
        for (int e = 0; e < 16; ++e) {
            int kb = kbase + hi * 16 + e;              // 16-bit B 32x16 layout
            b[e] = (_Float16)W[kb * GCNH + col0 + n];
        }
        c = __builtin_amdgcn_wmma_f32_16x16x32_f16(false, a, false, b,
                                                   (short)0, c, false, false);
    }
    #pragma unroll
    for (int r = 0; r < 8; ++r) {                      // C f32 16x16 layout
        C[(long long)(tile_m * 16 + r + hi * 8) * GCNH + col0 + n] = c[r];
    }
}

// ---------------------------------------------------------------------------
// Layer-2 aggregation + epilogue + mean pool, fully fused, zero f32 atomics
// in the hot loop. 16 threads per node (float4 feature chunks); per-edge
// metadata broadcast within the group, xw gathers coalesce to 256B lines.
// pooled[f] += relu( dinv^2*xw[n] + sum w*xw[s] + b2 )
// ---------------------------------------------------------------------------
__global__ void gather64_pool_kernel(const int* __restrict__ offsets, const int* __restrict__ counts,
                                     const int2* __restrict__ csr,
                                     const float* __restrict__ xw,
                                     const float* __restrict__ dinv,
                                     const float* __restrict__ b2,
                                     float* __restrict__ pooled, int n) {
    __shared__ float sums[GCNH];
    int tid = threadIdx.x;
    int f4  = tid & 15;                 // float4 chunk of 64 feats
    int g   = tid >> 4;                 // node group 0..15
    if (tid < GCNH) sums[tid] = 0.0f;
    __syncthreads();

    const int fb = f4 * 4;
    const float4 bias = *(const float4*)(b2 + fb);
    float4 pacc = {0.f, 0.f, 0.f, 0.f};

    for (int node = blockIdx.x * 16 + g; node < n; node += gridDim.x * 16) {
        float di = dinv[node];
        float sl = di * di;
        const float4 xv = *(const float4*)(xw + (long long)node * GCNH + fb);
        float4 acc;
        acc.x = sl * xv.x; acc.y = sl * xv.y; acc.z = sl * xv.z; acc.w = sl * xv.w;
        int start = offsets[node];
        int cnt   = counts[node];
        for (int k = 0; k < cnt; ++k) {
            int2 ew = csr[start + k];
            float w = __int_as_float(ew.y);
            const float4 v = *(const float4*)(xw + (long long)ew.x * GCNH + fb);
            acc.x += w * v.x; acc.y += w * v.y; acc.z += w * v.z; acc.w += w * v.w;
        }
        pacc.x += fmaxf(acc.x + bias.x, 0.0f);
        pacc.y += fmaxf(acc.y + bias.y, 0.0f);
        pacc.z += fmaxf(acc.z + bias.z, 0.0f);
        pacc.w += fmaxf(acc.w + bias.w, 0.0f);
    }
    atomicAdd(&sums[fb + 0], pacc.x);   // LDS atomics (per block)
    atomicAdd(&sums[fb + 1], pacc.y);
    atomicAdd(&sums[fb + 2], pacc.z);
    atomicAdd(&sums[fb + 3], pacc.w);
    __syncthreads();
    if (tid < GCNH) unsafeAtomicAdd(&pooled[tid], sums[tid]);
}

// ---------------------------------------------------------------------------
// GRU over T pooled vectors + final FC. One block, 128 threads, sequential T.
// ---------------------------------------------------------------------------
__global__ void gru_fc_kernel(const float* __restrict__ pooled,   // [T,64] (sums, pre-mean)
                              const float* __restrict__ w_ih,     // [384,64]
                              const float* __restrict__ w_hh,     // [384,128]
                              const float* __restrict__ b_ih, const float* __restrict__ b_hh,
                              const float* __restrict__ fc_w, const float* __restrict__ fc_b,
                              float* __restrict__ out, float invN) {
    __shared__ float xs[GCNH];
    __shared__ float hs[GRUH];
    __shared__ float red[GRUH];
    int j = threadIdx.x;                 // 0..127
    hs[j] = 0.0f;
    __syncthreads();
    for (int t = 0; t < TSTEPS; ++t) {
        if (j < GCNH) xs[j] = pooled[t * GCNH + j] * invN;
        __syncthreads();
        float gr = b_ih[j], gz = b_ih[GRUH + j], gn = b_ih[2 * GRUH + j];
        #pragma unroll 4
        for (int k = 0; k < GCNH; ++k) {
            float xk = xs[k];
            gr += xk * w_ih[j * GCNH + k];
            gz += xk * w_ih[(GRUH + j) * GCNH + k];
            gn += xk * w_ih[(2 * GRUH + j) * GCNH + k];
        }
        float hr = b_hh[j], hz = b_hh[GRUH + j], hn = b_hh[2 * GRUH + j];
        #pragma unroll 4
        for (int k = 0; k < GRUH; ++k) {
            float hk = hs[k];
            hr += hk * w_hh[j * GRUH + k];
            hz += hk * w_hh[(GRUH + j) * GRUH + k];
            hn += hk * w_hh[(2 * GRUH + j) * GRUH + k];
        }
        float rg = 1.0f / (1.0f + __expf(-(gr + hr)));
        float zg = 1.0f / (1.0f + __expf(-(gz + hz)));
        float ng = tanhf(gn + rg * hn);
        float hnew = (1.0f - zg) * ng + zg * hs[j];
        __syncthreads();
        hs[j] = hnew;
        __syncthreads();
    }
    red[j] = hs[j] * fc_w[j];
    __syncthreads();
    for (int s = GRUH / 2; s > 0; s >>= 1) {
        if (j < s) red[j] += red[j + s];
        __syncthreads();
    }
    if (j == 0) out[0] = red[0] + fc_b[0];
}

// ---------------------------------------------------------------------------
// Launcher
// ---------------------------------------------------------------------------
extern "C" void kernel_launch(void* const* d_in, const int* in_sizes, int n_in,
                              void* d_out, int out_size, void* d_ws, size_t ws_size,
                              hipStream_t stream) {
    const float*     x_seq = (const float*)d_in[0];        // [T,N,3]
    const long long* eidx  = (const long long*)d_in[1];    // [2,E] int64
    const float*     W1    = (const float*)d_in[2];        // [3,64]
    const float*     b1    = (const float*)d_in[3];        // [64]
    const float*     W2    = (const float*)d_in[4];        // [64,64]
    const float*     b2    = (const float*)d_in[5];        // [64]
    const float*     w_ih  = (const float*)d_in[6];        // [384,64]
    const float*     w_hh  = (const float*)d_in[7];        // [384,128]
    const float*     b_ih  = (const float*)d_in[8];
    const float*     b_hh  = (const float*)d_in[9];
    const float*     fc_w  = (const float*)d_in[10];       // [1,128]
    const float*     fc_b  = (const float*)d_in[11];       // [1]
    float*           out   = (float*)d_out;

    const int E = in_sizes[1] / 2;
    const int N = in_sizes[0] / (TSTEPS * 3);
    const long long NF = (long long)N * GCNH;

    const long long* dst64 = eidx + E;

    // ---- workspace layout (256B aligned) ----
    auto align_up = [](size_t v) { return (v + 255) & ~(size_t)255; };
    char*  ws      = (char*)d_ws;
    size_t off     = 0;
    int*   counts  = (int*)  (ws + off); off += align_up((size_t)N * 4);
    int*   offsets = (int*)  (ws + off); off += align_up((size_t)N * 4);
    int*   cursor  = (int*)  (ws + off); off += align_up((size_t)N * 4);
    int*   bsums   = (int*)  (ws + off); off += align_up((size_t)SCAN_B * 4);
    float* dinv    = (float*)(ws + off); off += align_up((size_t)N * 4);
    int2*  csr     = (int2*) (ws + off); off += align_up((size_t)E * 8);
    float* aggx    = (float*)(ws + off); off += align_up((size_t)N * 3 * 4);
    float* xw      = (float*)(ws + off); off += align_up((size_t)NF * 4);
    float* h1      = (float*)(ws + off); off += align_up((size_t)NF * 4);
    float* pooled  = (float*)(ws + off); off += align_up((size_t)TSTEPS * GCNH * 4);
    (void)ws_size;

    const int nBlocks  = (N + 255) / 256;
    const int eBlocks  = (E + 255) / 256;
    const int nScanBlk = (N + SCAN_B - 1) / SCAN_B;        // 196 for N=50000 (<= SCAN_B)

    // ---- once per call: CSR + dinv ----
    hipMemsetAsync(counts, 0, (size_t)N * 4, stream);
    count_kernel<<<eBlocks, 256, 0, stream>>>(dst64, counts, E);
    scan1_kernel<<<nScanBlk, SCAN_B, 0, stream>>>(counts, offsets, bsums, N);
    scan2_kernel<<<1, SCAN_B, 0, stream>>>(bsums, nScanBlk);
    scan3_kernel<<<nScanBlk, SCAN_B, 0, stream>>>(offsets, bsums, N);
    dinv_kernel<<<nBlocks, 256, 0, stream>>>(counts, dinv, N);
    copy_kernel<<<nBlocks, 256, 0, stream>>>(offsets, cursor, N);
    fill_kernel<<<eBlocks, 256, 0, stream>>>(eidx, dinv, cursor, csr, E);

    hipMemsetAsync(pooled, 0, (size_t)TSTEPS * GCNH * 4, stream);

    const int nfBlocks   = (int)((NF + 255) / 256);
    const int tiles      = (N / 16) * (GCNH / 16);
    const int gemmBlocks = (tiles + 7) / 8;                // 8 waves / 256-thread block

    for (int t = 0; t < TSTEPS; ++t) {
        const float* xt = x_seq + (long long)t * N * 3;

        // ---- GCN layer 1: rank-3 gather, then W1 expansion ----
        gather3_kernel<<<nBlocks, 256, 0, stream>>>(offsets, counts, csr, xt, dinv, aggx, N);
        epi1_kernel<<<nfBlocks, 256, 0, stream>>>(aggx, W1, b1, h1, N);

        // ---- GCN layer 2: WMMA linear, then fused gather + relu + pool ----
        gemm64_wmma_kernel<<<gemmBlocks, 256, 0, stream>>>(h1, W2, xw, N);
        gather64_pool_kernel<<<512, 256, 0, stream>>>(offsets, counts, csr, xw, dinv,
                                                      b2, pooled + t * GCNH, N);
    }

    // ---- GRU + FC ----
    gru_fc_kernel<<<1, GRUH, 0, stream>>>(pooled, w_ih, w_hh, b_ih, b_hh,
                                          fc_w, fc_b, out, 1.0f / (float)N);
}
